// LocalAttention_26603027432182
// MI455X (gfx1250) — matrix operand order repaired and checked by reading, not compiled, not run on hardware
//
#include <hip/hip_runtime.h>
#include <hip/hip_bf16.h>

typedef __attribute__((ext_vector_type(16))) _Float16 v16h;
typedef __attribute__((ext_vector_type(8)))  float    v8f;
typedef __attribute__((ext_vector_type(4)))  unsigned int u32x4;
typedef __attribute__((ext_vector_type(8)))  int          i32x8;
typedef __attribute__((ext_vector_type(4)))  int          i32x4;

#define S_LEN   4096
#define DHEAD   64
#define HALFW   50
#define KVSTR   68   // f32 stride for K/V LDS rows (64 data + 4 pad dwords)
#define PSTR    34   // f16 stride for P LDS rows (pad 32 -> 34)

#if defined(__gfx1250__) && __has_builtin(__builtin_amdgcn_tensor_load_to_lds) && \
    __has_builtin(__builtin_amdgcn_s_wait_tensorcnt)
#define USE_TDM 1
#else
#define USE_TDM 0
#endif

#if USE_TDM
// Issue a TDM 2-D tile load: 32 rows x 64 f32 from row-major global (stride 64)
// into LDS at lds_off, inserting 4 pad dwords after every 64 (-> 68 f32 rows).
__device__ __forceinline__ void tdm_load_chunk(const float* gptr, unsigned lds_off)
{
    const unsigned long long ga = (unsigned long long)(const void*)gptr;
    u32x4 g0;
    g0[0] = 1u;                                   // count=1 (valid descriptor)
    g0[1] = lds_off;                              // lds_addr (bytes)
    g0[2] = (unsigned)(ga & 0xFFFFFFFFu);         // global_addr[31:0]
    g0[3] = (unsigned)((ga >> 32) & 0x01FFFFFFu)  // global_addr[56:32]
          | (2u << 30);                           // type=2 ("image")
    i32x8 g1;
    g1[0] = (int)((2u << 16)      // data_size = 4 bytes
                | (1u << 20)      // pad_enable
                | (5u << 22)      // pad_interval: 2^(5+1)=64 dwords
                | (3u << 25));    // pad_amount: 3+1=4 dwords
    g1[1] = (int)(64u << 16);     // tensor_dim0[15:0]=64 (bits 63:48)
    g1[2] = (int)(32u << 16);     // tensor_dim1[15:0]=32 (bits 95:80)
    g1[3] = (int)(64u << 16);     // tile_dim0=64       (bits 127:112)
    g1[4] = 32;                   // tile_dim1=32       (bits 143:128)
    g1[5] = 64;                   // tensor_dim0_stride=64 (bits 191:160)
    g1[6] = 0;
    g1[7] = 0;
    i32x4 z4 = {0, 0, 0, 0};                 // groups 2/3 unused (2-D tensor)
    i32x8 z8 = {0, 0, 0, 0, 0, 0, 0, 0};     // extra descriptor group (unused)
    __builtin_amdgcn_tensor_load_to_lds(g0, g1, z4, z4, z8, 0);
}
#endif

__launch_bounds__(32)
__global__ void LocalAttention_26603027432182_kernel(
    const float* __restrict__ Q,
    const float* __restrict__ K,
    const float* __restrict__ V,
    float* __restrict__ O)
{
    __shared__ __align__(16) float     ldsK[2][32 * KVSTR];
    __shared__ __align__(16) float     ldsV[2][32 * KVSTR];
    __shared__ __align__(16) _Float16  ldsP[16 * PSTR];

    const int lane = threadIdx.x;          // 0..31 (wave32)
    const int g    = lane >> 4;            // lane group 0/1
    const int ln   = lane & 15;

    // Defensive: compiler cannot prove these LDS arrays are written (TDM writes
    // are invisible to alias analysis), so give it a may-write it can't kill.
    if ((int)blockDim.x == 0) { ldsK[0][0] = 0.f; ldsV[0][0] = 0.f; }

    const int ntiles = S_LEN / 16;
    const int qt = blockIdx.x % ntiles;
    const int bh = blockIdx.x / ntiles;
    const int q0 = qt * 16;
    const size_t base = (size_t)bh * S_LEN * DHEAD;
    const float* Qb = Q + base;
    const float* Kb = K + base;
    const float* Vb = V + base;
    float*       Ob = O + base;

    // ---- Load Q tile (16x64) as two 16x32 f16 A-fragments (kept in VGPRs) ----
    // A layout (16-bit 16x32): lane holds row M=ln; VGPR v half p -> K =
    // (v&3)*2 + p + 8*g + 16*(v>>2)
    v16h aQ0, aQ1;
    {
        const float* qrow = Qb + (size_t)(q0 + ln) * DHEAD;
#pragma unroll
        for (int v = 0; v < 8; ++v) {
            const int d = ((v & 3) << 1) + (g << 3) + ((v >> 2) << 4);
            aQ0[2 * v + 0] = (_Float16)qrow[d + 0];
            aQ0[2 * v + 1] = (_Float16)qrow[d + 1];
            aQ1[2 * v + 0] = (_Float16)qrow[32 + d + 0];
            aQ1[2 * v + 1] = (_Float16)qrow[32 + d + 1];
        }
    }

    // ---- Flash-attention running state (per row r+8g, replicated over 16 lanes) ----
    float m[8], l[8];
    v8f   oacc[4];
#pragma unroll
    for (int r = 0; r < 8; ++r) { m[r] = -1.0e30f; l[r] = 0.0f; }
#pragma unroll
    for (int db = 0; db < 4; ++db) {
        v8f z = {};
        oacc[db] = z;
    }

    int kmin = q0 - HALFW;        if (kmin < 0) kmin = 0;
    int kmax = q0 + 15 + HALFW;   if (kmax > S_LEN - 1) kmax = S_LEN - 1;
    const int cbeg = kmin & ~31;  // 32-aligned chunk start; all chunks stay in [0, S)

    const float scale = 0.125f;   // 1/sqrt(64)

#if USE_TDM
    const unsigned ldsK0 = (unsigned)(unsigned long long)(const void*)&ldsK[0][0];
    const unsigned ldsK1 = (unsigned)(unsigned long long)(const void*)&ldsK[1][0];
    const unsigned ldsV0 = (unsigned)(unsigned long long)(const void*)&ldsV[0][0];
    const unsigned ldsV1 = (unsigned)(unsigned long long)(const void*)&ldsV[1][0];
    // Prologue: DMA chunk 0 into buffer 0.
    tdm_load_chunk(Kb + (size_t)cbeg * DHEAD, ldsK0);
    tdm_load_chunk(Vb + (size_t)cbeg * DHEAD, ldsV0);
#endif

    int cur = 0;
    for (int c = cbeg; c <= kmax; c += 32, cur ^= 1) {
#if USE_TDM
        // Pipeline: issue next chunk's DMA into the other buffer, then wait
        // until only those 2 transfers remain outstanding (current chunk done).
        if (c + 32 <= kmax) {
            tdm_load_chunk(Kb + (size_t)(c + 32) * DHEAD, cur ? ldsK0 : ldsK1);
            tdm_load_chunk(Vb + (size_t)(c + 32) * DHEAD, cur ? ldsV0 : ldsV1);
            __builtin_amdgcn_s_wait_tensorcnt((short)2);
        } else {
            __builtin_amdgcn_s_wait_tensorcnt((short)0);
        }
#else
        // Fallback: stage K and V chunks (32 rows x 64 f32) into LDS, coalesced.
#pragma unroll
        for (int i = 0; i < 16; ++i) {
            const int idx  = i * 32 + lane;        // 0..511 float4 slots
            const int row  = idx >> 4;             // 0..31
            const int col4 = (idx & 15) << 2;      // 0,4,...,60
            const float4 kk = *(const float4*)(Kb + (size_t)(c + row) * DHEAD + col4);
            const float4 vv = *(const float4*)(Vb + (size_t)(c + row) * DHEAD + col4);
            *(float4*)(&ldsK[cur][row * KVSTR + col4]) = kk;
            *(float4*)(&ldsV[cur][row * KVSTR + col4]) = vv;
        }
#endif
        __syncthreads();

        const float* __restrict__ curK = &ldsK[cur][0];
        const float* __restrict__ curV = &ldsV[cur][0];

        // ---- Build K^T B-fragments and compute S = Q*K^T (16x32 scores) ----
        // B layout (16-bit 32x16): lane holds col N=ln; VGPR v half p -> K = (v&3)*2+p+8g+16*(v>>2)
        // B[d][n] = Key[c + j*16 + n][h*32 + d]
        v16h bK00, bK01, bK10, bK11;
#pragma unroll
        for (int v = 0; v < 8; ++v) {
            const int d = ((v & 3) << 1) + (g << 3) + ((v >> 2) << 4);
            const float* k0 = &curK[(0 * 16 + ln) * KVSTR];
            const float* k1 = &curK[(1 * 16 + ln) * KVSTR];
            bK00[2 * v + 0] = (_Float16)k0[d + 0];
            bK00[2 * v + 1] = (_Float16)k0[d + 1];
            bK01[2 * v + 0] = (_Float16)k0[32 + d + 0];
            bK01[2 * v + 1] = (_Float16)k0[32 + d + 1];
            bK10[2 * v + 0] = (_Float16)k1[d + 0];
            bK10[2 * v + 1] = (_Float16)k1[d + 1];
            bK11[2 * v + 0] = (_Float16)k1[32 + d + 0];
            bK11[2 * v + 1] = (_Float16)k1[32 + d + 1];
        }

        v8f cz = {};
        v8f s0 = __builtin_amdgcn_wmma_f32_16x16x32_f16(false, aQ0, false, bK00,
                                                        (short)0, cz, false, false);
        s0     = __builtin_amdgcn_wmma_f32_16x16x32_f16(false, aQ1, false, bK01,
                                                        (short)0, s0, false, false);
        v8f s1 = __builtin_amdgcn_wmma_f32_16x16x32_f16(false, aQ0, false, bK10,
                                                        (short)0, cz, false, false);
        s1     = __builtin_amdgcn_wmma_f32_16x16x32_f16(false, aQ1, false, bK11,
                                                        (short)0, s1, false, false);

        // ---- Online softmax over this 32-key chunk ----
        // Score element s{0,1}[r] is (row q0+r+8g, key c+{0,16}+ln).
        float alpha[8];
#pragma unroll
        for (int r = 0; r < 8; ++r) {
            const int qi = q0 + r + (g << 3);
            float x0 = s0[r] * scale;
            float x1 = s1[r] * scale;
            const int d0 = qi - (c + ln);
            const int d1 = qi - (c + 16 + ln);
            if (d0 < -HALFW || d0 > HALFW) x0 = -1.0e30f;
            if (d1 < -HALFW || d1 > HALFW) x1 = -1.0e30f;

            float mx = fmaxf(x0, x1);
            mx = fmaxf(mx, __shfl_xor(mx, 1, 32));
            mx = fmaxf(mx, __shfl_xor(mx, 2, 32));
            mx = fmaxf(mx, __shfl_xor(mx, 4, 32));
            mx = fmaxf(mx, __shfl_xor(mx, 8, 32));

            const float mnew = fmaxf(m[r], mx);
            const float a    = __expf(m[r] - mnew);
            const float p0   = __expf(x0 - mnew);
            const float p1   = __expf(x1 - mnew);
            float rs = p0 + p1;
            rs += __shfl_xor(rs, 1, 32);
            rs += __shfl_xor(rs, 2, 32);
            rs += __shfl_xor(rs, 4, 32);
            rs += __shfl_xor(rs, 8, 32);

            l[r] = l[r] * a + rs;
            m[r] = mnew;
            alpha[r] = a;

            // Spill P to LDS (row-major 16x32 f16) for the C->A layout transpose.
            ldsP[(r + (g << 3)) * PSTR + ln]      = (_Float16)p0;
            ldsP[(r + (g << 3)) * PSTR + 16 + ln] = (_Float16)p1;
        }
        __syncthreads();

        // ---- Read P back as a 16x32 A-fragment ----
        v16h aP;
#pragma unroll
        for (int v = 0; v < 8; ++v) {
            const int kk = ((v & 3) << 1) + (g << 3) + ((v >> 2) << 4);
            aP[2 * v + 0] = ldsP[ln * PSTR + kk + 0];
            aP[2 * v + 1] = ldsP[ln * PSTR + kk + 1];
        }

        // ---- O = O*alpha + P*V : V chunk as four 32x16 B-fragments ----
#pragma unroll
        for (int db = 0; db < 4; ++db) {
            v16h bV;
#pragma unroll
            for (int v = 0; v < 8; ++v) {
                const int vr = ((v & 3) << 1) + (g << 3) + ((v >> 2) << 4);
                bV[2 * v + 0] = (_Float16)curV[(vr + 0) * KVSTR + db * 16 + ln];
                bV[2 * v + 1] = (_Float16)curV[(vr + 1) * KVSTR + db * 16 + ln];
            }
            v8f co;
#pragma unroll
            for (int r = 0; r < 8; ++r) co[r] = oacc[db][r] * alpha[r];
            oacc[db] = __builtin_amdgcn_wmma_f32_16x16x32_f16(false, aP, false, bV,
                                                              (short)0, co, false, false);
        }
        __syncthreads();   // done with this buffer before it is re-targeted
    }

    // ---- Epilogue: normalize by l and store (row q0+r+8g, col db*16+ln) ----
#pragma unroll
    for (int db = 0; db < 4; ++db) {
        const int dcol = db * 16 + ln;
#pragma unroll
        for (int r = 0; r < 8; ++r) {
            const float inv = 1.0f / l[r];
            Ob[(size_t)(q0 + r + (g << 3)) * DHEAD + dcol] = oacc[db][r] * inv;
        }
    }
}

extern "C" void kernel_launch(void* const* d_in, const int* in_sizes, int n_in,
                              void* d_out, int out_size, void* d_ws, size_t ws_size,
                              hipStream_t stream) {
    (void)in_sizes; (void)n_in; (void)out_size; (void)d_ws; (void)ws_size;
    const float* Q = (const float*)d_in[0];
    const float* K = (const float*)d_in[1];
    const float* V = (const float*)d_in[2];
    float*       O = (float*)d_out;

    const int BH     = 2 * 8;
    const int ntiles = S_LEN / 16;
    dim3 grid(BH * ntiles);   // 4096 blocks
    dim3 block(32);           // one wave32 per 16-query tile
    LocalAttention_26603027432182_kernel<<<grid, block, 0, stream>>>(Q, K, V, O);
}
